// LingMess_model_35038343201531
// MI455X (gfx1250) — compile-verified, matrix operand order checked by reading, not run on hardware
//
#include <hip/hip_runtime.h>
#include <math.h>

// ---------------- problem constants (match reference) ----------------
#define S_LEN 2048
#define B_SZ  2
#define HIDD  1024
#define NCATS 7
#define NF    (NCATS*1024)
#define MAXK  819
#define KPAD  1024
#define SPAN  30
#define BANDW 32
#define NEGV  (-10000.0f)

typedef __attribute__((ext_vector_type(16))) __bf16 v16bf;
typedef __attribute__((ext_vector_type(8)))  float  v8f;

__device__ __forceinline__ unsigned short f2bf(float f) {
  unsigned u = __float_as_uint(f);
  unsigned r = u + 0x7FFFu + ((u >> 16) & 1u);   // round-to-nearest-even
  return (unsigned short)(r >> 16);
}
__device__ __forceinline__ float bf2f(unsigned short h) {
  return __uint_as_float(((unsigned)h) << 16);
}

// ---------------- elementwise f32 -> bf16 convert ----------------
__global__ void cvt_kernel(const float* __restrict__ in, unsigned short* __restrict__ out, long n) {
  long i = (long)blockIdx.x * 256 + threadIdx.x;
  long stride = (long)gridDim.x * 256;
  for (; i < n; i += stride) out[i] = f2bf(in[i]);
}

// transpose-convert: in [ncats, F, G] f32 -> out [ncats, G, F] bf16 (F=G=1024)
__global__ void cvtT_kernel(const float* __restrict__ in, unsigned short* __restrict__ out,
                            int F, int G) {
  __shared__ float t[32][33];
  int n = blockIdx.z;
  int g0 = blockIdx.x * 32, f0 = blockIdx.y * 32;
  int tx = threadIdx.x & 31, ty = threadIdx.x >> 5;   // 32 x 8
  const float* src = in + (size_t)n * F * G;
  unsigned short* dst = out + (size_t)n * F * G;
#pragma unroll
  for (int rr = 0; rr < 4; ++rr) {
    int f = f0 + ty + rr * 8;
    t[ty + rr * 8][tx] = src[(size_t)f * G + g0 + tx];
  }
  __syncthreads();
#pragma unroll
  for (int rr = 0; rr < 4; ++rr) {
    int g = g0 + ty + rr * 8;
    dst[(size_t)g * F + f0 + tx] = f2bf(t[tx][ty + rr * 8]);
  }
}

// ---------------- WMMA bf16 GEMM: C[M,N] = A[M,K] @ B[N,K]^T ----------------
// CDNA5 path: double-buffered LDS staged via global_load_async_to_lds_b128 (ASYNCcnt),
// 8x v_wmma_f32_16x16x32_bf16 per k-step per wave.
// epilogue: (+= C if doAcc) + bias[n] -> exact GELU (opt) -> store f32 C and optional bf16 Cb
#define TM 128
#define TN 128
#define KB 32
#define LDST 40   // padded LDS row stride (elements)

union AFrag { v16bf v; uint4 u[2]; };

__device__ __forceinline__ unsigned lds_addr_of(const void* p) {
  // low 32 bits of a generic pointer to LDS == LDS byte offset (CDNA5 aperture rule)
  return (unsigned)(unsigned long long)p;
}
__device__ __forceinline__ void async_cp16(unsigned lds_b, unsigned long long gaddr) {
  asm volatile("global_load_async_to_lds_b128 %0, %1, off"
               :: "v"(lds_b), "v"(gaddr) : "memory");
}
__device__ __forceinline__ void wait_async0() {
  asm volatile("s_wait_asynccnt 0x0" ::: "memory");
}

__global__ __launch_bounds__(256) void gemm_bf16(
    const unsigned short* __restrict__ A, const unsigned short* __restrict__ Bm,
    float* __restrict__ C, unsigned short* __restrict__ Cb,
    const float* __restrict__ bias, int M, int N, int K, int doGelu, int doAcc)
{
  __shared__ unsigned short As[2][TM * LDST];
  __shared__ unsigned short Bs[2][TN * LDST];
  int tid = threadIdx.x;
  int lane = tid & 31, w = tid >> 5;
  int wm = (w & 3) * 32, wn = (w >> 2) * 64;          // 4x2 wave grid, 32x64 per wave
  int mblk = blockIdx.y * TM, nblk = blockIdx.x * TN;
  int r16 = lane & 15, khalf = lane >> 4;

  // Branch-free staging: clamp OOB rows (garbage only feeds rows/cols masked at store).
  int c0 = tid, c1 = tid + 256;                       // chunk ids 0..511 (128 rows x 4 chunks)
  int rA0 = c0 >> 2, kA0 = (c0 & 3) * 8;
  int rA1 = c1 >> 2, kA1 = (c1 & 3) * 8;
  int gA0 = min(mblk + rA0, M - 1), gA1 = min(mblk + rA1, M - 1);
  int gB0 = min(nblk + rA0, N - 1), gB1 = min(nblk + rA1, N - 1);
  unsigned long long pA0 = (unsigned long long)(A + (size_t)gA0 * K + kA0);
  unsigned long long pA1 = (unsigned long long)(A + (size_t)gA1 * K + kA1);
  unsigned long long pB0 = (unsigned long long)(Bm + (size_t)gB0 * K + kA0);
  unsigned long long pB1 = (unsigned long long)(Bm + (size_t)gB1 * K + kA1);
  unsigned lA0[2], lA1[2], lB0[2], lB1[2];
#pragma unroll
  for (int bu = 0; bu < 2; ++bu) {
    lA0[bu] = lds_addr_of(&As[bu][rA0 * LDST + kA0]);
    lA1[bu] = lds_addr_of(&As[bu][rA1 * LDST + kA1]);
    lB0[bu] = lds_addr_of(&Bs[bu][rA0 * LDST + kA0]);
    lB1[bu] = lds_addr_of(&Bs[bu][rA1 * LDST + kA1]);
  }

  v8f zero = {0.f,0.f,0.f,0.f,0.f,0.f,0.f,0.f};
  v8f acc[2][4];
#pragma unroll
  for (int i = 0; i < 2; ++i)
#pragma unroll
    for (int j = 0; j < 4; ++j) acc[i][j] = zero;

  const int nk = K / KB;
  // prologue: stage tile 0 into buffer 0
  async_cp16(lA0[0], pA0); async_cp16(lA1[0], pA1);
  async_cp16(lB0[0], pB0); async_cp16(lB1[0], pB1);
  wait_async0();
  __syncthreads();

  for (int it = 0; it < nk; ++it) {
    int nxt = it + 1;
    if (nxt < nk) {
      // issue async copies for tile it+1 into the alternate buffer (RAW covered by
      // end-of-iteration wait+barrier; WAR covered by previous iteration's barrier)
      unsigned long long koff = (unsigned long long)(2 * nxt * KB);
      int bu = nxt & 1;
      async_cp16(lA0[bu], pA0 + koff); async_cp16(lA1[bu], pA1 + koff);
      async_cp16(lB0[bu], pB0 + koff); async_cp16(lB1[bu], pB1 + koff);
    }
    // CDNA5 16-bit fragment layout: lanes 0-15 rows, K chunks {0..7,16..23};
    // lanes 16-31 same rows, K chunks {8..15,24..31}
    const unsigned short* as_ = As[it & 1];
    const unsigned short* bs_ = Bs[it & 1];
    AFrag af[2], bfr[4];
#pragma unroll
    for (int i = 0; i < 2; ++i) {
      int row = wm + i * 16 + r16;
      af[i].u[0] = *(const uint4*)(as_ + row * LDST + khalf * 8);
      af[i].u[1] = *(const uint4*)(as_ + row * LDST + 16 + khalf * 8);
    }
#pragma unroll
    for (int j = 0; j < 4; ++j) {
      int row = wn + j * 16 + r16;
      bfr[j].u[0] = *(const uint4*)(bs_ + row * LDST + khalf * 8);
      bfr[j].u[1] = *(const uint4*)(bs_ + row * LDST + 16 + khalf * 8);
    }
#pragma unroll
    for (int i = 0; i < 2; ++i)
#pragma unroll
      for (int j = 0; j < 4; ++j)
        acc[i][j] = __builtin_amdgcn_wmma_f32_16x16x32_bf16(
            false, af[i].v, false, bfr[j].v, (short)0, acc[i][j], false, false);
    wait_async0();
    __syncthreads();
  }

  // C tile layout: lane holds column N = lane%16; VGPR r -> row r + 8*(lane>=16)
  int rowoff = khalf * 8;
#pragma unroll
  for (int i = 0; i < 2; ++i) {
#pragma unroll
    for (int j = 0; j < 4; ++j) {
      int col = nblk + wn + j * 16 + r16;
      if (col >= N) continue;
      float bv = bias ? bias[col] : 0.0f;
#pragma unroll
      for (int r = 0; r < 8; ++r) {
        int row = mblk + wm + i * 16 + rowoff + r;
        if (row >= M) continue;
        size_t idx = (size_t)row * N + col;
        float x = acc[i][j][r];
        if (doAcc) x += C[idx];
        x += bv;
        if (doGelu) x = 0.5f * x * (1.0f + erff(x * 0.70710678118654752f));
        C[idx] = x;
        if (Cb) Cb[idx] = f2bf(x);
      }
    }
  }
}

// ---------------- rowwise LayerNorm (+optional vector-dot) ----------------
// permK==0: out[row*F+f]; permK=K: row=(b*K+k), f=(n*1024+ff) -> out[((b*7+n)*K+k)*1024+ff]
__global__ __launch_bounds__(256) void ln_kernel(
    const float* __restrict__ h, int F, const float* __restrict__ g, const float* __restrict__ beta,
    unsigned short* __restrict__ outb, int permK,
    const float* __restrict__ dotw, const float* __restrict__ dotb, float* __restrict__ dotout)
{
  __shared__ float red[256], red2[256];
  int row = blockIdx.x, tid = threadIdx.x;
  const float* hr = h + (size_t)row * F;
  float s0 = 0.f, s1 = 0.f;
  for (int f = tid; f < F; f += 256) { float x = hr[f]; s0 += x; s1 += x * x; }
  red[tid] = s0; red2[tid] = s1; __syncthreads();
  for (int off = 128; off > 0; off >>= 1) {
    if (tid < off) { red[tid] += red[tid + off]; red2[tid] += red2[tid + off]; }
    __syncthreads();
  }
  float mu = red[0] / F;
  float var = red2[0] / F - mu * mu;
  float inv = rsqrtf(var + 1e-5f);
  float d = 0.f;
  for (int f = tid; f < F; f += 256) {
    float x = (hr[f] - mu) * inv * g[f] + beta[f];
    size_t dst;
    if (permK) {
      int b = row / permK, k = row % permK;
      int n = f >> 10, ff = f & 1023;
      dst = (((size_t)(b * NCATS + n)) * permK + k) * 1024 + ff;
    } else dst = (size_t)row * F + f;
    outb[dst] = f2bf(x);
    if (dotw) d += x * dotw[f];
  }
  if (dotw) {
    __syncthreads();
    red[tid] = d; __syncthreads();
    for (int off = 128; off > 0; off >>= 1) {
      if (tid < off) red[tid] += red[tid + off];
      __syncthreads();
    }
    if (tid == 0) dotout[row] = red[0] + dotb[0];
  }
}

// ---------------- banded joint scores: only 0 <= t-s < 30 can win top-k ----------------
__global__ __launch_bounds__(256) void band_kernel(
    const unsigned short* __restrict__ p, const unsigned short* __restrict__ e,
    const float* __restrict__ smv, const float* __restrict__ emv, float* __restrict__ band)
{
  __shared__ float red[256];
  int bs = blockIdx.x;                 // b*S + s
  int b = bs >> 11, s = bs & (S_LEN - 1);
  int tid = threadIdx.x;
  const unsigned short* pr = p + (size_t)bs * 1024;
  float pv[4];
#pragma unroll
  for (int i = 0; i < 4; ++i) pv[i] = bf2f(pr[tid + 256 * i]);
  float sm = smv[bs];
  for (int j = 0; j < BANDW; ++j) {
    int t = s + j;
    int valid = (j < SPAN) && (t < S_LEN);
    int tc = valid ? t : s;
    const unsigned short* er = e + ((size_t)b * S_LEN + tc) * 1024;
    float d = 0.f;
#pragma unroll
    for (int i = 0; i < 4; ++i) d += pv[i] * bf2f(er[tid + 256 * i]);
    red[tid] = d; __syncthreads();
    for (int off = 128; off > 0; off >>= 1) {
      if (tid < off) red[tid] += red[tid + off];
      __syncthreads();
    }
    if (tid == 0)
      band[(size_t)bs * BANDW + j] = valid ? (red[0] + sm + emv[b * S_LEN + tc]) : -1e30f;
    __syncthreads();
  }
}

// ---------------- deterministic top-K + ascending index sort (one WG per batch) ----------------
__device__ __forceinline__ unsigned fkey(float v) {
  unsigned u = __float_as_uint(v);
  return (u & 0x80000000u) ? ~u : (u | 0x80000000u);
}

__global__ __launch_bounds__(1024) void topk_kernel(
    const float* __restrict__ band, int* __restrict__ selidx, float* __restrict__ selval)
{
  __shared__ unsigned redc[1024];
  __shared__ int sidx[1024];
  __shared__ float sval[1024];
  __shared__ unsigned acnt;
  const int NQ = S_LEN * BANDW;
  int b = blockIdx.x, tid = threadIdx.x;
  const float* bb = band + (size_t)b * NQ;

  // binary search for the K-th largest sortable key T
  unsigned lo = 0u, hi = 0xFFFFFFFFu;
  for (int it = 0; it < 33 && lo < hi; ++it) {
    unsigned mid = lo + ((hi - lo) >> 1) + 1u;
    unsigned c = 0;
    for (int q = tid; q < NQ; q += 1024) c += (fkey(bb[q]) >= mid) ? 1u : 0u;
    redc[tid] = c; __syncthreads();
    for (int off = 512; off > 0; off >>= 1) {
      if (tid < off) redc[tid] += redc[tid + off];
      __syncthreads();
    }
    unsigned tot = redc[0]; __syncthreads();
    if (tot >= MAXK) lo = mid; else hi = mid - 1u;
  }
  unsigned T = lo;

  unsigned c = 0;
  for (int q = tid; q < NQ; q += 1024) c += (fkey(bb[q]) > T) ? 1u : 0u;
  redc[tid] = c; __syncthreads();
  for (int off = 512; off > 0; off >>= 1) {
    if (tid < off) redc[tid] += redc[tid + off];
    __syncthreads();
  }
  unsigned Cgt = redc[0]; __syncthreads();

  sidx[tid] = 0x7FFFFFFF; sval[tid] = 0.f;
  if (tid == 0) acnt = 0u;
  __syncthreads();

  for (int q = tid; q < NQ; q += 1024) {
    float v = bb[q];
    unsigned k = fkey(v);
    int s = q >> 5, j = q & (BANDW - 1);
    int flat = s * S_LEN + s + j;
    if (k > T) {
      unsigned pp = atomicAdd(&acnt, 1u);
      sidx[pp] = flat; sval[pp] = v;
    } else if (k == T) {                 // deterministic tie-break by source index
      unsigned r = 0;
      for (int q2 = 0; q2 < q; ++q2) if (fkey(bb[q2]) == T) ++r;
      unsigned slot = Cgt + r;
      if (slot < MAXK) { sidx[slot] = flat; sval[slot] = v; }
    }
  }
  __syncthreads();

  // bitonic sort ascending by flat index (unique keys -> deterministic)
  for (int k2 = 2; k2 <= 1024; k2 <<= 1)
    for (int j2 = k2 >> 1; j2 > 0; j2 >>= 1) {
      int ixj = tid ^ j2;
      if (ixj > tid) {
        bool up = ((tid & k2) == 0);
        int a = sidx[tid], bq = sidx[ixj];
        if ((a > bq) == up) {
          float va = sval[tid], vb = sval[ixj];
          sidx[tid] = bq; sidx[ixj] = a;
          sval[tid] = vb; sval[ixj] = va;
        }
      }
      __syncthreads();
    }
  selidx[b * KPAD + tid] = sidx[tid];
  selval[b * KPAD + tid] = sval[tid];
}

// ---------------- gather span reps -> bf16 ----------------
__global__ void gather_kernel(const float* __restrict__ emb, const int* __restrict__ selidx,
                              int useEnd, unsigned short* __restrict__ out)
{
  long total = (long)B_SZ * MAXK * 1024;
  for (long i = (long)blockIdx.x * 256 + threadIdx.x; i < total; i += (long)gridDim.x * 256) {
    long row = i >> 10; int f = (int)(i & 1023);
    int b = (int)(row / MAXK), k = (int)(row % MAXK);
    int idx = selidx[b * KPAD + k];
    int pos = useEnd ? (idx % S_LEN) : (idx / S_LEN);
    out[i] = f2bf(emb[((size_t)b * S_LEN + pos) * 1024 + f]);
  }
}

// ---------------- bias vector: a_s.(Bs2s+Be2s) + a_e.(Be2e+Bs2e) per (b,n,k) ----------------
__global__ __launch_bounds__(256) void biasdot_kernel(
    const unsigned short* __restrict__ as_, const unsigned short* __restrict__ ae_,
    const float* __restrict__ Bs2s, const float* __restrict__ Be2e,
    const float* __restrict__ Bs2e, const float* __restrict__ Be2s,
    float* __restrict__ biasvec)
{
  __shared__ float red[256];
  int r = blockIdx.x, tid = threadIdx.x;
  int k = r % MAXK; int n = (r / MAXK) % NCATS; int b = r / (MAXK * NCATS);
  size_t base = (((size_t)(b * NCATS + n)) * MAXK + k) * 1024;
  float d = 0.f;
  for (int f = tid; f < 1024; f += 256) {
    d += bf2f(as_[base + f]) * (Bs2s[n * 1024 + f] + Be2s[n * 1024 + f]);
    d += bf2f(ae_[base + f]) * (Be2e[n * 1024 + f] + Bs2e[n * 1024 + f]);
  }
  red[tid] = d; __syncthreads();
  for (int off = 128; off > 0; off >>= 1) {
    if (tid < off) red[tid] += red[tid + off];
    __syncthreads();
  }
  if (tid == 0) biasvec[r] = red[0];
}

// ---------------- final in-place epilogue on d_out ----------------
__global__ void final_kernel(float* __restrict__ out, const float* __restrict__ biasvec,
                             const float* __restrict__ selval)
{
  long total = (long)B_SZ * NCATS * MAXK * MAXK;
  for (long i = (long)blockIdx.x * 256 + threadIdx.x; i < total; i += (long)gridDim.x * 256) {
    int l = (int)(i % MAXK);
    long t = i / MAXK;
    int k = (int)(t % MAXK);
    long t2 = t / MAXK;
    int n = (int)(t2 % NCATS);
    int b = (int)(t2 / NCATS);
    float x = out[i];
    x += biasvec[(b * NCATS + n) * MAXK + l];
    x += selval[b * KPAD + k] + selval[b * KPAD + l];
    if (!(l < k)) x += NEGV;               // strictly-lower-triangular mask
    out[i] = x;
  }
}

// ---------------- host orchestration ----------------
static inline void gemm(const unsigned short* A, const unsigned short* B, float* C,
                        unsigned short* Cb, const float* bias, int M, int N, int K,
                        int doGelu, int doAcc, hipStream_t st) {
  dim3 grid((N + TN - 1) / TN, (M + TM - 1) / TM);
  gemm_bf16<<<grid, 256, 0, st>>>(A, B, C, Cb, bias, M, N, K, doGelu, doAcc);
}

extern "C" void kernel_launch(void* const* d_in, const int* in_sizes, int n_in,
                              void* d_out, int out_size, void* d_ws, size_t ws_size,
                              hipStream_t stream) {
  (void)in_sizes; (void)n_in; (void)out_size; (void)ws_size;
  const float* input_emb = (const float*)d_in[0];
  const float* sm_W  = (const float*)d_in[2];  const float* sm_b  = (const float*)d_in[3];
  const float* sm_g  = (const float*)d_in[4];  const float* sm_be = (const float*)d_in[5];
  const float* em_W  = (const float*)d_in[6];  const float* em_b  = (const float*)d_in[7];
  const float* em_g  = (const float*)d_in[8];  const float* em_be = (const float*)d_in[9];
  const float* ms_w  = (const float*)d_in[10]; const float* ms_b  = (const float*)d_in[11];
  const float* me_w  = (const float*)d_in[12]; const float* me_b  = (const float*)d_in[13];
  const float* s2e_W = (const float*)d_in[14]; const float* s2e_b = (const float*)d_in[15];
  const float* cs_W  = (const float*)d_in[16]; const float* cs_b  = (const float*)d_in[17];
  const float* cs_g  = (const float*)d_in[18]; const float* cs_be = (const float*)d_in[19];
  const float* ce_W  = (const float*)d_in[20]; const float* ce_b  = (const float*)d_in[21];
  const float* ce_g  = (const float*)d_in[22]; const float* ce_be = (const float*)d_in[23];
  const float* Ws2s  = (const float*)d_in[24]; const float* We2e  = (const float*)d_in[25];
  const float* Ws2e  = (const float*)d_in[26]; const float* We2s  = (const float*)d_in[27];
  const float* Bs2s  = (const float*)d_in[28]; const float* Be2e  = (const float*)d_in[29];
  const float* Bs2e  = (const float*)d_in[30]; const float* Be2s  = (const float*)d_in[31];

  char* w = (char*)d_ws;
  auto take = [&](size_t bytes) -> char* {
    char* p = w; w += (bytes + 255) & ~(size_t)255; return p;
  };
  const size_t MM = (size_t)1024 * 1024;
  unsigned short* embB  = (unsigned short*)take(2ull * B_SZ * S_LEN * 1024);
  unsigned short* smWB  = (unsigned short*)take(2ull * MM);
  unsigned short* emWB  = (unsigned short*)take(2ull * MM);
  unsigned short* s2eWB = (unsigned short*)take(2ull * MM);
  unsigned short* csWB  = (unsigned short*)take(2ull * NF * 1024);
  unsigned short* ceWB  = (unsigned short*)take(2ull * NF * 1024);
  unsigned short* WtS2S = (unsigned short*)take(2ull * NCATS * MM);
  unsigned short* WtE2E = (unsigned short*)take(2ull * NCATS * MM);
  unsigned short* WtS2E = (unsigned short*)take(2ull * NCATS * MM);
  unsigned short* WtE2S = (unsigned short*)take(2ull * NCATS * MM);
  float* hbuf = (float*)take(4ull * (size_t)(B_SZ * MAXK) * NF);   // covers 4096x1024 too
  unsigned short* sB = (unsigned short*)take(2ull * B_SZ * S_LEN * 1024);
  unsigned short* eB = (unsigned short*)take(2ull * B_SZ * S_LEN * 1024);
  unsigned short* pB = (unsigned short*)take(2ull * B_SZ * S_LEN * 1024);
  float* smv = (float*)take(4ull * B_SZ * S_LEN);
  float* emv = (float*)take(4ull * B_SZ * S_LEN);
  float* band = (float*)take(4ull * B_SZ * S_LEN * BANDW);
  int*   selidx = (int*)take(4ull * B_SZ * KPAD);
  float* selval = (float*)take(4ull * B_SZ * KPAD);
  unsigned short* srB = (unsigned short*)take(2ull * B_SZ * MAXK * 1024);
  unsigned short* erB = (unsigned short*)take(2ull * B_SZ * MAXK * 1024);
  unsigned short* asB = (unsigned short*)take(2ull * B_SZ * NCATS * MAXK * 1024);
  unsigned short* aeB = (unsigned short*)take(2ull * B_SZ * NCATS * MAXK * 1024);
  float* biasvec = (float*)take(4ull * B_SZ * NCATS * MAXK);
  float* Pf = (float*)take(4ull * MAXK * 1024);
  float* Qf = (float*)take(4ull * MAXK * 1024);
  unsigned short* PbB = (unsigned short*)take(2ull * MAXK * 1024);
  unsigned short* QbB = (unsigned short*)take(2ull * MAXK * 1024);

  // 1) convert operands to bf16 (weights once, activations per stage)
  cvt_kernel<<<2048, 256, 0, stream>>>(input_emb, embB, (long)B_SZ * S_LEN * 1024);
  cvt_kernel<<<1024, 256, 0, stream>>>(sm_W, smWB, (long)MM);
  cvt_kernel<<<1024, 256, 0, stream>>>(em_W, emWB, (long)MM);
  cvt_kernel<<<1024, 256, 0, stream>>>(s2e_W, s2eWB, (long)MM);
  cvt_kernel<<<2048, 256, 0, stream>>>(cs_W, csWB, (long)NF * 1024);
  cvt_kernel<<<2048, 256, 0, stream>>>(ce_W, ceWB, (long)NF * 1024);
  {
    dim3 tg(32, 32, NCATS);
    cvtT_kernel<<<tg, 256, 0, stream>>>(Ws2s, WtS2S, 1024, 1024);
    cvtT_kernel<<<tg, 256, 0, stream>>>(We2e, WtE2E, 1024, 1024);
    cvtT_kernel<<<tg, 256, 0, stream>>>(Ws2e, WtS2E, 1024, 1024);
    cvtT_kernel<<<tg, 256, 0, stream>>>(We2s, WtE2S, 1024, 1024);
  }

  const int MS = B_SZ * S_LEN;   // 4096
  // 2) mention MLPs: GEMM+GELU then LN (+fused mention-score dot)
  gemm(embB, smWB, hbuf, nullptr, sm_b, MS, 1024, 1024, 1, 0, stream);
  ln_kernel<<<MS, 256, 0, stream>>>(hbuf, 1024, sm_g, sm_be, sB, 0, ms_w, ms_b, smv);
  gemm(embB, emWB, hbuf, nullptr, em_b, MS, 1024, 1024, 1, 0, stream);
  ln_kernel<<<MS, 256, 0, stream>>>(hbuf, 1024, em_g, em_be, eB, 0, me_w, me_b, emv);

  // 3) p = s @ s2e_W^T + s2e_b   (bf16 secondary output)
  gemm(sB, s2eWB, hbuf, pB, s2e_b, MS, 1024, 1024, 0, 0, stream);

  // 4) band scores + top-k span pruning
  band_kernel<<<MS, 256, 0, stream>>>(pB, eB, smv, emv, band);
  topk_kernel<<<B_SZ, 1024, 0, stream>>>(band, selidx, selval);

  // 5) gather + per-category coref MLPs (LN writes [B,7,K,1024] bf16 directly)
  gather_kernel<<<2048, 256, 0, stream>>>(input_emb, selidx, 0, srB);
  gather_kernel<<<2048, 256, 0, stream>>>(input_emb, selidx, 1, erB);
  gemm(srB, csWB, hbuf, nullptr, cs_b, B_SZ * MAXK, NF, 1024, 1, 0, stream);
  ln_kernel<<<B_SZ * MAXK, 256, 0, stream>>>(hbuf, NF, cs_g, cs_be, asB, MAXK,
                                             nullptr, nullptr, nullptr);
  gemm(erB, ceWB, hbuf, nullptr, ce_b, B_SZ * MAXK, NF, 1024, 1, 0, stream);
  ln_kernel<<<B_SZ * MAXK, 256, 0, stream>>>(hbuf, NF, ce_g, ce_be, aeB, MAXK,
                                             nullptr, nullptr, nullptr);

  // 6) bias vectors
  biasdot_kernel<<<B_SZ * NCATS * MAXK, 256, 0, stream>>>(asB, aeB, Bs2s, Be2e, Bs2e, Be2s, biasvec);

  // 7) bilinear: P=a_s@Ws2s+a_e@We2s, Q=a_e@We2e+a_s@Ws2e, logits=P@a_s^T+Q@a_e^T -> d_out
  float* out = (float*)d_out;
  for (int b = 0; b < B_SZ; ++b)
    for (int n = 0; n < NCATS; ++n) {
      const unsigned short* asp = asB + (((size_t)(b * NCATS + n)) * MAXK) * 1024;
      const unsigned short* aep = aeB + (((size_t)(b * NCATS + n)) * MAXK) * 1024;
      gemm(asp, WtS2S + (size_t)n * MM, Pf, nullptr, nullptr, MAXK, 1024, 1024, 0, 0, stream);
      gemm(aep, WtE2S + (size_t)n * MM, Pf, PbB, nullptr, MAXK, 1024, 1024, 0, 1, stream);
      gemm(aep, WtE2E + (size_t)n * MM, Qf, nullptr, nullptr, MAXK, 1024, 1024, 0, 0, stream);
      gemm(asp, WtS2E + (size_t)n * MM, Qf, QbB, nullptr, MAXK, 1024, 1024, 0, 1, stream);
      float* op = out + ((size_t)(b * NCATS + n)) * MAXK * MAXK;
      gemm(PbB, asp, op, nullptr, nullptr, MAXK, MAXK, 1024, 0, 0, stream);
      gemm(QbB, aep, op, nullptr, nullptr, MAXK, MAXK, 1024, 0, 1, stream);
    }

  // 8) in-place epilogue: bias + pair scores + triangular NEG mask
  final_kernel<<<4096, 256, 0, stream>>>(out, biasvec, selval);
}
